// EdgeMessage_38878043963415
// MI455X (gfx1250) — matrix operand order repaired and checked by reading, not compiled, not run on hardware
//
#include <hip/hip_runtime.h>

typedef __attribute__((ext_vector_type(16))) _Float16 v16h;
typedef __attribute__((ext_vector_type(8)))  _Float16 v8h;
typedef __attribute__((ext_vector_type(8)))  float    v8f;

constexpr int N_NODES = 50000;
constexpr long N_EDGES = 1600000;
constexpr int HID = 64;

// Fragment counts (ktiles*4 ntiles) per matrix, 512 f16 per fragment.
constexpr int EFRAGS = (6 + 2 + 2) * 4;   // 40  -> eW1|eW2|eW3 blob
constexpr int NFRAGS = (4 + 2 + 2) * 4;   // 32  -> nW1|nW2|nW3 blob

// B-fragment element for V_WMMA_F32_16X16X32_F16 (ISA 7.12.2): fragment
// (ktile,ntile); lanes 0-15 hold K=kt*32+{0..7,16..23} of column n=lane&15,
// lanes 16-31 hold K=kt*32+{8..15,24..31}. Stored [frag][lane][16] so each lane's 16 f16
// are 32B contiguous (two ds_load_b128 on read-back).
__device__ __forceinline__ _Float16 frag_elem(const float* __restrict__ W, int f) {
  const int frag  = f >> 9;
  const int lane  = (f >> 4) & 31;
  const int i     = f & 15;
  const int ktile = frag >> 2;
  const int ntile = frag & 3;
  const int k = ktile * 32 + ((lane >> 4) << 3) + (i & 7) + ((i >> 3) << 4);
  const int n = (ntile << 4) + (lane & 15);
  return (_Float16)W[k * 64 + n];
}

// One-shot: swizzle all weight matrices into f16 WMMA-B fragment blobs.
__global__ __launch_bounds__(256) void prep_weights_kernel(
    const float* __restrict__ eW1, const float* __restrict__ eW2,
    const float* __restrict__ eW3, const float* __restrict__ nW1,
    const float* __restrict__ nW2, const float* __restrict__ nW3,
    _Float16* __restrict__ eblob, _Float16* __restrict__ nblob) {
  const int idx = blockIdx.x * 256 + threadIdx.x;
  const int E1 = 24 * 512, E2 = 8 * 512, E3 = 8 * 512;
  const int N1 = 16 * 512, N2 = 8 * 512, N3 = 8 * 512;
  if (idx < E1)                eblob[idx] = frag_elem(eW1, idx);
  else if (idx < E1 + E2)      eblob[idx] = frag_elem(eW2, idx - E1);
  else if (idx < E1 + E2 + E3) eblob[idx] = frag_elem(eW3, idx - E1 - E2);
  else {
    const int j = idx - (E1 + E2 + E3);
    if (j < N1)                nblob[j] = frag_elem(nW1, j);
    else if (j < N1 + N2)      nblob[j] = frag_elem(nW2, j - N1);
    else if (j < N1 + N2 + N3) nblob[j] = frag_elem(nW3, j - N1 - N2);
  }
}

// Bulk global->LDS staging via CDNA5 async copy (ASYNCcnt-tracked).
// GVS mode: mem_addr = SADDR64 + VADDR32; VDST VGPR holds LDS byte offset.
// Generic shared pointers carry the LDS offset in addr[31:0] (ISA 10.2
// aperture table), so truncation yields the DS offset.
__device__ __forceinline__ void async_stage_wblob(const _Float16* __restrict__ gsrc,
                                                  _Float16* __restrict__ ldst,
                                                  int n_chunks_per_thread, int tid) {
  const unsigned lds0 = (unsigned)(unsigned long long)(void*)ldst;
  const unsigned long long g0 = (unsigned long long)(const void*)gsrc;
  for (int it = 0; it < n_chunks_per_thread; ++it) {
    const unsigned off = (unsigned)(it * 4096 + tid * 16);
    asm volatile("global_load_async_to_lds_b128 %0, %1, %2"
                 :: "v"(lds0 + off), "v"(off), "s"(g0)
                 : "memory");
  }
  asm volatile("s_wait_asynccnt 0x0" ::: "memory");
}

__device__ __forceinline__ v16h ldsA(const _Float16* __restrict__ hrow, int kb) {
  // A-fragment (16-bit 16x32): lane (m,hs): elems 0..7 = K kb..kb+7,
  // elems 8..15 = K kb+16..kb+23 of row m (kb includes hs*8).
  v8h lo = *(const v8h*)(hrow + kb);
  v8h hi = *(const v8h*)(hrow + kb + 16);
  v16h out;
  __builtin_memcpy(&out, &lo, 16);
  __builtin_memcpy(((char*)&out) + 16, &hi, 16);
  return out;
}

// ---------------------------------------------------------------- edge MLP ---
__global__ __launch_bounds__(256) void edge_mlp_kernel(
    const float* __restrict__ node_feat, const float* __restrict__ edge_feat,
    const int* __restrict__ edge_idx, const _Float16* __restrict__ wblob,
    const float* __restrict__ b1, const float* __restrict__ b2,
    const float* __restrict__ b3, const float* __restrict__ gam,
    const float* __restrict__ bet,
    float* __restrict__ out_edge, float* __restrict__ nodal) {
  __shared__ __align__(16) _Float16 sW[EFRAGS * 512];     // 40 KB
  __shared__ __align__(32) _Float16 sH[8][16 * 64];       // 16 KB
  __shared__ float sB1[64], sB2[64], sB3[64], sG[64], sBt[64];

  const int tid = threadIdx.x;
  async_stage_wblob(wblob, sW, EFRAGS * 512 * 2 / (256 * 16), tid);  // 10 chunks
  if (tid < 64) {
    sB1[tid] = b1[tid]; sB2[tid] = b2[tid]; sB3[tid] = b3[tid];
    sG[tid] = gam[tid]; sBt[tid] = bet[tid];
  }
  __syncthreads();

  const _Float16* sW1 = sW;
  const _Float16* sW2 = sW + 24 * 512;
  const _Float16* sW3 = sW + 32 * 512;

  const int wave = tid >> 5;
  const int lane = tid & 31;
  const int m    = lane & 15;
  const int hs   = lane >> 4;
  const long base = (long)blockIdx.x * 128 + wave * 16;
  const long e    = base + m;
  const long eg   = (e < N_EDGES) ? e : 0;
  const int ic = edge_idx[eg * 2 + 0];
  const int is = edge_idx[eg * 2 + 1];

  // Gather x = [edge_feat | node[c] | node[s]] directly into A fragments.
  _Float16 xa[6][16];
#pragma unroll
  for (int t = 0; t < 6; ++t) {
#pragma unroll
    for (int g = 0; g < 2; ++g) {
      const int k0 = t * 32 + hs * 8 + g * 16;  // 8-aligned, one segment
      const float* src;
      if (k0 < 64)       src = edge_feat + eg * 64 + k0;
      else if (k0 < 128) src = node_feat + (long)ic * 64 + (k0 - 64);
      else               src = node_feat + (long)is * 64 + (k0 - 128);
      const float4 u0 = ((const float4*)src)[0];
      const float4 u1 = ((const float4*)src)[1];
      _Float16* d = &xa[t][g * 8];
      d[0] = (_Float16)u0.x; d[1] = (_Float16)u0.y; d[2] = (_Float16)u0.z; d[3] = (_Float16)u0.w;
      d[4] = (_Float16)u1.x; d[5] = (_Float16)u1.y; d[6] = (_Float16)u1.z; d[7] = (_Float16)u1.w;
    }
  }

  // ---- layer 1: 16x192 @ 192x64, relu ----
#pragma unroll
  for (int nt = 0; nt < 4; ++nt) {
    v8f acc = {};
#pragma unroll
    for (int t = 0; t < 6; ++t) {
      v16h av; __builtin_memcpy(&av, xa[t], 32);
      v16h bv = *(const v16h*)&sW1[(t * 4 + nt) * 512 + lane * 16];
      acc = __builtin_amdgcn_wmma_f32_16x16x32_f16(false, av, false, bv, (short)0, acc, false, false);
    }
    const int n = nt * 16 + m;
    const float bb = sB1[n];
#pragma unroll
    for (int r = 0; r < 8; ++r) {
      float v = acc[r] + bb;
      v = v > 0.f ? v : 0.f;
      sH[wave][(r + hs * 8) * 64 + n] = (_Float16)v;
    }
  }
  __syncthreads();
  v16h h1a[2];
#pragma unroll
  for (int t = 0; t < 2; ++t) h1a[t] = ldsA(&sH[wave][m * 64], t * 32 + hs * 8);
  __syncthreads();

  // ---- layer 2: 16x64 @ 64x64, relu ----
#pragma unroll
  for (int nt = 0; nt < 4; ++nt) {
    v8f acc = {};
#pragma unroll
    for (int t = 0; t < 2; ++t) {
      v16h bv = *(const v16h*)&sW2[(t * 4 + nt) * 512 + lane * 16];
      acc = __builtin_amdgcn_wmma_f32_16x16x32_f16(false, h1a[t], false, bv, (short)0, acc, false, false);
    }
    const int n = nt * 16 + m;
    const float bb = sB2[n];
#pragma unroll
    for (int r = 0; r < 8; ++r) {
      float v = acc[r] + bb;
      v = v > 0.f ? v : 0.f;
      sH[wave][(r + hs * 8) * 64 + n] = (_Float16)v;
    }
  }
  __syncthreads();
  v16h h2a[2];
#pragma unroll
  for (int t = 0; t < 2; ++t) h2a[t] = ldsA(&sH[wave][m * 64], t * 32 + hs * 8);

  // ---- layer 3: 16x64 @ 64x64 + bias (no relu) ----
  v8f h3[4];
#pragma unroll
  for (int nt = 0; nt < 4; ++nt) {
    v8f acc = {};
#pragma unroll
    for (int t = 0; t < 2; ++t) {
      v16h bv = *(const v16h*)&sW3[(t * 4 + nt) * 512 + lane * 16];
      acc = __builtin_amdgcn_wmma_f32_16x16x32_f16(false, h2a[t], false, bv, (short)0, acc, false, false);
    }
    const float bb = sB3[nt * 16 + m];
#pragma unroll
    for (int r = 0; r < 8; ++r) acc[r] = acc[r] + bb;
    h3[nt] = acc;
  }

  // ---- layernorm per row, write out, segment-sum atomics ----
#pragma unroll
  for (int r = 0; r < 8; ++r) {
    const float x0 = h3[0][r], x1 = h3[1][r], x2 = h3[2][r], x3 = h3[3][r];
    float sm = x0 + x1 + x2 + x3;
    float sq = x0 * x0 + x1 * x1 + x2 * x2 + x3 * x3;
#pragma unroll
    for (int msk = 8; msk >= 1; msk >>= 1) {
      sm += __shfl_xor(sm, msk, 32);
      sq += __shfl_xor(sq, msk, 32);
    }
    const float mean = sm * (1.f / 64.f);
    const float var  = sq * (1.f / 64.f) - mean * mean;
    const float inv  = rsqrtf(var + 1e-5f);
    const int row = r + hs * 8;
    const int crow = __shfl(ic, row, 32);   // center id of edge in this row
    const long eo = base + row;
    if (eo < N_EDGES) {
#pragma unroll
      for (int nt = 0; nt < 4; ++nt) {
        const int n = nt * 16 + m;
        const float y = (h3[nt][r] - mean) * inv * sG[n] + sBt[n];
        out_edge[eo * 64 + n] = y;
        __hip_atomic_fetch_add(&nodal[(long)crow * 64 + n], y,
                               __ATOMIC_RELAXED, __HIP_MEMORY_SCOPE_AGENT);
      }
    }
  }
}

// ---------------------------------------------------------------- node MLP ---
__global__ __launch_bounds__(256) void node_mlp_kernel(
    const float* __restrict__ node_feat, const float* __restrict__ nodal,
    const _Float16* __restrict__ wblob,
    const float* __restrict__ b1, const float* __restrict__ b2,
    const float* __restrict__ b3, const float* __restrict__ gam,
    const float* __restrict__ bet, float* __restrict__ out_node) {
  __shared__ __align__(16) _Float16 sW[NFRAGS * 512];     // 32 KB
  __shared__ __align__(32) _Float16 sH[8][16 * 64];
  __shared__ float sB1[64], sB2[64], sB3[64], sG[64], sBt[64];

  const int tid = threadIdx.x;
  async_stage_wblob(wblob, sW, NFRAGS * 512 * 2 / (256 * 16), tid);  // 8 chunks
  if (tid < 64) {
    sB1[tid] = b1[tid]; sB2[tid] = b2[tid]; sB3[tid] = b3[tid];
    sG[tid] = gam[tid]; sBt[tid] = bet[tid];
  }
  __syncthreads();

  const _Float16* sW1 = sW;
  const _Float16* sW2 = sW + 16 * 512;
  const _Float16* sW3 = sW + 24 * 512;

  const int wave = tid >> 5;
  const int lane = tid & 31;
  const int m    = lane & 15;
  const int hs   = lane >> 4;
  const long base = (long)blockIdx.x * 128 + wave * 16;
  const long v    = base + m;
  const long vg   = (v < N_NODES) ? v : 0;

  _Float16 xa[4][16];
#pragma unroll
  for (int t = 0; t < 4; ++t) {
#pragma unroll
    for (int g = 0; g < 2; ++g) {
      const int k0 = t * 32 + hs * 8 + g * 16;
      const float* src = (k0 < 64) ? (node_feat + vg * 64 + k0)
                                   : (nodal + vg * 64 + (k0 - 64));
      const float4 u0 = ((const float4*)src)[0];
      const float4 u1 = ((const float4*)src)[1];
      _Float16* d = &xa[t][g * 8];
      d[0] = (_Float16)u0.x; d[1] = (_Float16)u0.y; d[2] = (_Float16)u0.z; d[3] = (_Float16)u0.w;
      d[4] = (_Float16)u1.x; d[5] = (_Float16)u1.y; d[6] = (_Float16)u1.z; d[7] = (_Float16)u1.w;
    }
  }

#pragma unroll
  for (int nt = 0; nt < 4; ++nt) {
    v8f acc = {};
#pragma unroll
    for (int t = 0; t < 4; ++t) {
      v16h av; __builtin_memcpy(&av, xa[t], 32);
      v16h bv = *(const v16h*)&sW1[(t * 4 + nt) * 512 + lane * 16];
      acc = __builtin_amdgcn_wmma_f32_16x16x32_f16(false, av, false, bv, (short)0, acc, false, false);
    }
    const int n = nt * 16 + m;
    const float bb = sB1[n];
#pragma unroll
    for (int r = 0; r < 8; ++r) {
      float x = acc[r] + bb;
      x = x > 0.f ? x : 0.f;
      sH[wave][(r + hs * 8) * 64 + n] = (_Float16)x;
    }
  }
  __syncthreads();
  v16h h1a[2];
#pragma unroll
  for (int t = 0; t < 2; ++t) h1a[t] = ldsA(&sH[wave][m * 64], t * 32 + hs * 8);
  __syncthreads();

#pragma unroll
  for (int nt = 0; nt < 4; ++nt) {
    v8f acc = {};
#pragma unroll
    for (int t = 0; t < 2; ++t) {
      v16h bv = *(const v16h*)&sW2[(t * 4 + nt) * 512 + lane * 16];
      acc = __builtin_amdgcn_wmma_f32_16x16x32_f16(false, h1a[t], false, bv, (short)0, acc, false, false);
    }
    const int n = nt * 16 + m;
    const float bb = sB2[n];
#pragma unroll
    for (int r = 0; r < 8; ++r) {
      float x = acc[r] + bb;
      x = x > 0.f ? x : 0.f;
      sH[wave][(r + hs * 8) * 64 + n] = (_Float16)x;
    }
  }
  __syncthreads();
  v16h h2a[2];
#pragma unroll
  for (int t = 0; t < 2; ++t) h2a[t] = ldsA(&sH[wave][m * 64], t * 32 + hs * 8);

  v8f h3[4];
#pragma unroll
  for (int nt = 0; nt < 4; ++nt) {
    v8f acc = {};
#pragma unroll
    for (int t = 0; t < 2; ++t) {
      v16h bv = *(const v16h*)&sW3[(t * 4 + nt) * 512 + lane * 16];
      acc = __builtin_amdgcn_wmma_f32_16x16x32_f16(false, h2a[t], false, bv, (short)0, acc, false, false);
    }
    const float bb = sB3[nt * 16 + m];
#pragma unroll
    for (int r = 0; r < 8; ++r) acc[r] = acc[r] + bb;
    h3[nt] = acc;
  }

#pragma unroll
  for (int r = 0; r < 8; ++r) {
    const float x0 = h3[0][r], x1 = h3[1][r], x2 = h3[2][r], x3 = h3[3][r];
    float sm = x0 + x1 + x2 + x3;
    float sq = x0 * x0 + x1 * x1 + x2 * x2 + x3 * x3;
#pragma unroll
    for (int msk = 8; msk >= 1; msk >>= 1) {
      sm += __shfl_xor(sm, msk, 32);
      sq += __shfl_xor(sq, msk, 32);
    }
    const float mean = sm * (1.f / 64.f);
    const float var  = sq * (1.f / 64.f) - mean * mean;
    const float inv  = rsqrtf(var + 1e-5f);
    const long vo = base + r + hs * 8;
    if (vo < N_NODES) {
#pragma unroll
      for (int nt = 0; nt < 4; ++nt) {
        const int n = nt * 16 + m;
        out_node[vo * 64 + n] = (h3[nt][r] - mean) * inv * sG[n] + sBt[n];
      }
    }
  }
}

extern "C" void kernel_launch(void* const* d_in, const int* in_sizes, int n_in,
                              void* d_out, int out_size, void* d_ws, size_t ws_size,
                              hipStream_t stream) {
  const float* node_feat = (const float*)d_in[0];
  const float* edge_feat = (const float*)d_in[1];
  const int*   edge_idx  = (const int*)d_in[2];
  const float* eW1 = (const float*)d_in[4];
  const float* eb1 = (const float*)d_in[5];
  const float* eW2 = (const float*)d_in[6];
  const float* eb2 = (const float*)d_in[7];
  const float* eW3 = (const float*)d_in[8];
  const float* eb3 = (const float*)d_in[9];
  const float* eg  = (const float*)d_in[10];
  const float* ebt = (const float*)d_in[11];
  const float* nW1 = (const float*)d_in[12];
  const float* nb1 = (const float*)d_in[13];
  const float* nW2 = (const float*)d_in[14];
  const float* nb2 = (const float*)d_in[15];
  const float* nW3 = (const float*)d_in[16];
  const float* nb3 = (const float*)d_in[17];
  const float* ng  = (const float*)d_in[18];
  const float* nbt = (const float*)d_in[19];

  float* out_edge = (float*)d_out;
  float* out_node = (float*)d_out + (size_t)N_EDGES * HID;

  // Workspace layout: nodal accumulator | edge weight blob | node weight blob
  float* nodal = (float*)d_ws;
  _Float16* eblob = (_Float16*)((char*)d_ws + (size_t)N_NODES * HID * sizeof(float));
  _Float16* nblob = eblob + EFRAGS * 512;

  hipMemsetAsync(nodal, 0, (size_t)N_NODES * HID * sizeof(float), stream);

  const int prep_total = (EFRAGS + NFRAGS) * 512;          // 36864
  prep_weights_kernel<<<(prep_total + 255) / 256, 256, 0, stream>>>(
      eW1, eW2, eW3, nW1, nW2, nW3, eblob, nblob);

  const int edge_blocks = (int)(N_EDGES / 128);            // 12500, exact
  edge_mlp_kernel<<<edge_blocks, 256, 0, stream>>>(
      node_feat, edge_feat, edge_idx, eblob, eb1, eb2, eb3, eg, ebt,
      out_edge, nodal);

  const int node_blocks = (N_NODES + 127) / 128;           // 391
  node_mlp_kernel<<<node_blocks, 256, 0, stream>>>(
      node_feat, nodal, nblob, nb1, nb2, nb3, ng, nbt, out_node);
}
